// JungCNNDecoder_13005160972606
// MI455X (gfx1250) — matrix operand order repaired and checked by reading, not compiled, not run on hardware
//
#include <hip/hip_runtime.h>
#include <hip/hip_bf16.h>

// CDNA5 / gfx1250 fused CNN decoder, wave32.
// WMMA f32_16x16x32_f16 for conv1/conv2 (im2col GEMM) and fc1.
// K-dimension is kernel-position-major and activations are stored
// position-major in LDS, so ALL WMMA operand loads are aligned vector
// LDS/global loads (no scalar gathers, no predication).

typedef __attribute__((ext_vector_type(16))) _Float16 v16h;
typedef __attribute__((ext_vector_type(8)))  _Float16 v8h;
typedef __attribute__((ext_vector_type(8)))  float    v8f;
typedef __attribute__((ext_vector_type(4)))  int      v4i;

#define SB 8                    // samples per workgroup
#define NTHREADS 256            // 8 waves

// ---- workspace byte offsets (packed fp16 weights + folded BN params) ----
#define WS_W1E   0u             // [64][160] fp16, K = kp*16+ic   (20480 B)
#define WS_W2E   20480u         // [64][256] fp16, K = kp*64+ic   (32768 B)
#define WS_FC1E  53248u         // [64][5184] fp16, K = c*81+pos  (663552 B)
#define WS_BNP   716800u        // 4 x 64 f32: scale1, shift1, scale2, shift2

// ---- LDS byte offsets (all 16B aligned) ----
#define L_IDX    0u             // 48 ints: z_idx[24], x_idx[24]
#define L_BN     192u           // 256 f32
#define L_GRID   1216u          // [8][112 pos][16 ch] fp16, zero pad (28672 B)
#define L_ACT1   29888u         // [8][100 pos][64 ch] fp16, zero pad (102400 B)
#define L_ACT2   132288u        // [8][5184] fp16 (82944 B)
#define L_W1     215232u        // [64][160] fp16 (20480 B)
#define L_W2     235712u        // [64][256] fp16 (32768 B)
#define L_FCP    268480u        // [2][16][64] f32 fc1 K-partials (8192 B)
#define L_FCO    276672u        // [16][64] f32 (4096 B)
#define SMEM_BYTES 280768u

// --------------------------------------------------------------------------
// Kernel 0: repack weights fp32 -> fp16 kernel-position-major B layouts,
//           fold conv bias + BN into per-channel scale/shift.
// --------------------------------------------------------------------------
__global__ __launch_bounds__(256)
void jung_repack(const float* __restrict__ w1, const float* __restrict__ w2,
                 const float* __restrict__ fc1_w,
                 const float* __restrict__ b1, const float* __restrict__ g1,
                 const float* __restrict__ beta1, const float* __restrict__ m1,
                 const float* __restrict__ v1,
                 const float* __restrict__ b2, const float* __restrict__ g2,
                 const float* __restrict__ beta2, const float* __restrict__ m2,
                 const float* __restrict__ v2,
                 _Float16* __restrict__ w1e, _Float16* __restrict__ w2e,
                 _Float16* __restrict__ fc1e, float* __restrict__ bnp)
{
    const int idx = blockIdx.x * blockDim.x + threadIdx.x;
    const int N_W1 = 64 * 160;                 // K = kp*16+ic, kp 9 = zero pad
    const int N_W2 = 64 * 256;                 // K = kp*64+ic
    const int N_FC = 64 * 5184;                // N padded 50 -> 64
    const int total = N_W1 + N_W2 + N_FC + 256;
    if (idx >= total) return;

    if (idx < N_W1) {
        int oc = idx / 160, r = idx % 160;
        int kp = r >> 4, ic = r & 15;          // kp = ky*3+kx
        float v = 0.f;
        if (kp < 9) {
            int ky = kp / 3, kx = kp % 3;
            v = w1[((oc * 16 + ic) * 3 + ky) * 3 + kx];
        }
        w1e[idx] = (_Float16)v;
    } else if (idx < N_W1 + N_W2) {
        int i = idx - N_W1;
        int oc = i / 256, r = i % 256;
        int kp = r >> 6, ic = r & 63;          // kp = ky*2+kx
        int ky = kp >> 1, kx = kp & 1;
        w2e[i] = (_Float16)w2[((oc * 64 + ic) * 2 + ky) * 2 + kx];
    } else if (idx < N_W1 + N_W2 + N_FC) {
        int i = idx - (N_W1 + N_W2);
        int n = i / 5184, k = i % 5184;        // k = c*81 + pos (matches reshape)
        fc1e[i] = (_Float16)((n < 50) ? fc1_w[n * 5184 + k] : 0.f);
    } else {
        int t = idx - (N_W1 + N_W2 + N_FC);
        int c = t & 63, which = t >> 6;
        if (which == 0) {
            bnp[c] = g1[c] * rsqrtf(v1[c] + 1e-5f);
        } else if (which == 1) {
            float sc = g1[c] * rsqrtf(v1[c] + 1e-5f);
            bnp[64 + c] = (b1[c] - m1[c]) * sc + beta1[c];
        } else if (which == 2) {
            bnp[128 + c] = g2[c] * rsqrtf(v2[c] + 1e-5f);
        } else {
            float sc = g2[c] * rsqrtf(v2[c] + 1e-5f);
            bnp[192 + c] = (b2[c] - m2[c]) * sc + beta2[c];
        }
    }
}

// --------------------------------------------------------------------------
// Kernel 1: fused scatter + conv1 + conv2 + fc1 + fc2 for 8 samples per WG.
// --------------------------------------------------------------------------
__global__ __launch_bounds__(NTHREADS)
void jung_fused(const int* __restrict__ syndrome,
                const int* __restrict__ Hz, const int* __restrict__ Hx,
                const _Float16* __restrict__ w1e, const _Float16* __restrict__ w2e,
                const _Float16* __restrict__ fc1e, const float* __restrict__ bnp,
                const float* __restrict__ fc1_b, const float* __restrict__ fc2_w,
                const float* __restrict__ fc2_b, float* __restrict__ out)
{
    extern __shared__ char smem[];
    int*      idx_l  = (int*)(smem + L_IDX);
    float*    bn_l   = (float*)(smem + L_BN);
    _Float16* gridt  = (_Float16*)(smem + L_GRID);   // [8][112][16] pos-major
    _Float16* act1t  = (_Float16*)(smem + L_ACT1);   // [8][100][64] pos-major
    _Float16* act2_l = (_Float16*)(smem + L_ACT2);   // [8][5184]
    _Float16* w1_l   = (_Float16*)(smem + L_W1);     // [64][160]
    _Float16* w2_l   = (_Float16*)(smem + L_W2);     // [64][256]
    float*    fcp_l  = (float*)(smem + L_FCP);       // [2][16][64]
    float*    fco_l  = (float*)(smem + L_FCO);       // [16][64]

    const int tid  = threadIdx.x;
    const int lane = tid & 31;
    const int wave = tid >> 5;
    const int b0   = blockIdx.x * SB;
    const int mrow = lane & 15;
    const int hw   = lane >> 4;            // half-wave select
    const int koff = hw * 8;               // unused; A uses bksl like B
    const int bksl = hw * 16;              // B K-offset per half-wave

    // ---- phase 0: zero padded tensors, stage weights, stabilizer indices
    {
        v4i z4 = {0, 0, 0, 0};
        v4i* gz = (v4i*)gridt;
        for (int i = tid; i < 28672 / 16; i += NTHREADS) gz[i] = z4;
        v4i* az = (v4i*)act1t;
        for (int i = tid; i < 102400 / 16; i += NTHREADS) az[i] = z4;
    }
    if (tid < 48) {
        const int* H = (tid < 24) ? Hz : Hx;
        int row = (tid < 24) ? tid : tid - 24;
        int mr = 1 << 20, mc = 1 << 20;
        for (int q = 0; q < 49; ++q) {
            if (H[row * 49 + q] == 1) {
                int rq = q / 7, cq = q % 7;      // L = 7 qubit lattice
                mr = rq < mr ? rq : mr;
                mc = cq < mc ? cq : mc;
            }
        }
        idx_l[tid] = mr * 8 + mc;                // GRID = 8
    }
    bn_l[tid] = bnp[tid];
    {
        const int* s1 = (const int*)w1e; int* d1 = (int*)w1_l;
        for (int i = tid; i < 64 * 160 / 2; i += NTHREADS) d1[i] = s1[i];
        const int* s2 = (const int*)w2e; int* d2 = (int*)w2_l;
        for (int i = tid; i < 64 * 256 / 2; i += NTHREADS) d2[i] = s2[i];
    }
    __syncthreads();

    // interior of 8x8 = -0.5 for all 16 channels; border / pos>=100 stay 0
    for (int e = tid; e < SB * 64 * 16; e += NTHREADS) {
        int s = e >> 10, p = (e >> 4) & 63, ch = e & 15;   // ch contiguous
        int py = p >> 3, px = p & 7;
        gridt[(s * 112 + (py + 1) * 10 + (px + 1)) * 16 + ch] =
            (_Float16)(-0.5f);
    }
    __syncthreads();

    // scatter syndrome bits onto position-major grid
    for (int e = tid; e < SB * 8 * 48; e += NTHREADS) {
        int j = e % 48;
        int t = (e / 48) & 7;
        int s = e / (48 * 8);
        int v = syndrome[((b0 + s) * 8 + t) * 48 + j];
        int ch = (j < 24) ? (2 * t) : (2 * t + 1);
        int p  = idx_l[j];                       // idx_l[24..47] hold x_idx
        int py = p >> 3, px = p & 7;
        gridt[(s * 112 + (py + 1) * 10 + (px + 1)) * 16 + ch] =
            (_Float16)(float)v;
    }
    __syncthreads();

    // ---- phase 1: conv1 (3x3, 16->64ch, 8x8, pad 1): 64x64x160 GEMM/sample
    {
        const int s = wave;                      // one wave per sample
        for (int mt = 0; mt < 4; ++mt) {
            const int m  = mt * 16 + mrow;       // output position (A row)
            const int oy = m >> 3, ox = m & 7;
            const int pbase = s * 112 + oy * 10 + ox;   // +1 pad folded in
            v8f acc[4] = {};
#pragma unroll
            for (int ks = 0; ks < 5; ++ks) {
                const int kbase = ks * 32;
                const int klo = kbase + bksl;           // 8-run inside a block
                const int khi = kbase + bksl + 8;
                const int kplo = klo >> 4, iclo = klo & 15;
                const int kphi = khi >> 4, ichi = khi & 15;
                const int kylo = kplo / 3, kxlo = kplo - kylo * 3;
                const int kyhi = kphi / 3, kxhi = kphi - kyhi * 3;
                v8h alo = *(const v8h*)
                    &gridt[(pbase + kylo * 10 + kxlo) * 16 + iclo];
                v8h ahi = *(const v8h*)
                    &gridt[(pbase + kyhi * 10 + kxhi) * 16 + ichi];
                v16h a;
#pragma unroll
                for (int i = 0; i < 8; ++i) { a[i] = alo[i]; a[8 + i] = ahi[i]; }
#pragma unroll
                for (int nt = 0; nt < 4; ++nt) {
                    v16h b = *(const v16h*)
                        &w1_l[(nt * 16 + mrow) * 160 + kbase + bksl];
                    acc[nt] = __builtin_amdgcn_wmma_f32_16x16x32_f16(
                        false, a, false, b, (short)0, acc[nt], false, false);
                }
            }
#pragma unroll
            for (int nt = 0; nt < 4; ++nt) {
                const int n = nt * 16 + mrow;    // output channel (D col)
                const float sc = bn_l[n], sh = bn_l[64 + n];
#pragma unroll
                for (int r = 0; r < 8; ++r) {
                    int mm = mt * 16 + r + (hw << 3);
                    int py = mm >> 3, px = mm & 7;
                    float v = acc[nt][r] * sc + sh;
                    v = v > 0.f ? v : 0.f;
                    act1t[(s * 100 + (py + 1) * 10 + (px + 1)) * 64 + n] =
                        (_Float16)v;
                }
            }
        }
    }
    __syncthreads();

    // ---- phase 2: conv2 (2x2, 64->64ch, pad 1 -> 9x9): 96x64x256 GEMM/sample
    {
        const int s = wave;
        for (int mt = 0; mt < 6; ++mt) {
            const int m  = mt * 16 + mrow;
            const int mc = m < 81 ? m : 80;      // branchless clamp (rows 81+ junk)
            const int oy = mc / 9, ox = mc % 9;
            const int pbase = s * 100 + oy * 10 + ox;
            v8f acc[4] = {};
#pragma unroll
            for (int ks = 0; ks < 8; ++ks) {
                const int kbase = ks * 32;
                const int klo = kbase + bksl;
                const int khi = kbase + bksl + 8;
                const int kplo = klo >> 6, iclo = klo & 63;
                const int kphi = khi >> 6, ichi = khi & 63;
                v8h alo = *(const v8h*)
                    &act1t[(pbase + (kplo >> 1) * 10 + (kplo & 1)) * 64 + iclo];
                v8h ahi = *(const v8h*)
                    &act1t[(pbase + (kphi >> 1) * 10 + (kphi & 1)) * 64 + ichi];
                v16h a;
#pragma unroll
                for (int i = 0; i < 8; ++i) { a[i] = alo[i]; a[8 + i] = ahi[i]; }
#pragma unroll
                for (int nt = 0; nt < 4; ++nt) {
                    v16h b = *(const v16h*)
                        &w2_l[(nt * 16 + mrow) * 256 + kbase + bksl];
                    acc[nt] = __builtin_amdgcn_wmma_f32_16x16x32_f16(
                        false, a, false, b, (short)0, acc[nt], false, false);
                }
            }
#pragma unroll
            for (int nt = 0; nt < 4; ++nt) {
                const int n = nt * 16 + mrow;
                const float sc = bn_l[128 + n], sh = bn_l[192 + n];
#pragma unroll
                for (int r = 0; r < 8; ++r) {
                    int mm = mt * 16 + r + (hw << 3);
                    if (mm < 81) {
                        float v = acc[nt][r] * sc + sh;
                        v = v > 0.f ? v : 0.f;
                        act2_l[s * 5184 + n * 81 + mm] = (_Float16)v;
                    }
                }
            }
        }
    }
    __syncthreads();

    // ---- phase 3: fc1 (M=16 rows = 8 samples dup'd, N=64 pad, K=5184)
    // all 8 waves: wave = nt*2 + khalf; each does 81 of 162 K-steps
    {
        const int nt = wave >> 1;
        const int kh = wave & 1;
        const int n  = nt * 16 + mrow;
        const int sm = mrow & 7;                 // rows 8..15 mirror 0..7
        v8f acc = {};
        const int ks0 = kh * 81;
        for (int ks = ks0; ks < ks0 + 81; ++ks) {
            const int kbase = ks * 32;
            int pk = (ks + 8 < 162) ? (ks + 8) : 161;   // stream-ahead hint
            __builtin_prefetch(&fc1e[n * 5184 + pk * 32 + bksl], 0, 1);
            v8h alo = *(const v8h*)&act2_l[sm * 5184 + kbase + bksl];
            v8h ahi = *(const v8h*)&act2_l[sm * 5184 + kbase + bksl + 8];
            v16h a;
#pragma unroll
            for (int i = 0; i < 8; ++i) { a[i] = alo[i]; a[8 + i] = ahi[i]; }
            v16h b = *(const v16h*)&fc1e[n * 5184 + kbase + bksl];
            acc = __builtin_amdgcn_wmma_f32_16x16x32_f16(
                false, a, false, b, (short)0, acc, false, false);
        }
#pragma unroll
        for (int r = 0; r < 8; ++r) {
            int mm = r + (hw << 3);              // sample row
            fcp_l[(kh * 16 + mm) * 64 + n] = acc[r];
        }
    }
    __syncthreads();

    // combine K-halves + bias + relu
    for (int e = tid; e < 8 * 64; e += NTHREADS) {
        int row = e >> 6, col = e & 63;
        float v = fcp_l[row * 64 + col] + fcp_l[(16 + row) * 64 + col];
        v += (col < 50) ? fc1_b[col] : 0.f;
        v = v > 0.f ? v : 0.f;
        fco_l[row * 64 + col] = v;
    }
    __syncthreads();

    // ---- phase 4: fc2 (50 -> 4) per sample
    if (tid < 32) {
        int s = tid >> 2, j = tid & 3;
        float sum = fc2_b[j];
        for (int i = 0; i < 50; ++i)
            sum += fco_l[s * 64 + i] * fc2_w[j * 50 + i];
        out[(b0 + s) * 4 + j] = sum;
    }
}

// --------------------------------------------------------------------------
extern "C" void kernel_launch(void* const* d_in, const int* in_sizes, int n_in,
                              void* d_out, int out_size, void* d_ws, size_t ws_size,
                              hipStream_t stream) {
    const int*   syndrome = (const int*)d_in[0];
    const int*   Hz       = (const int*)d_in[1];
    const int*   Hx       = (const int*)d_in[2];
    const float* w1       = (const float*)d_in[3];
    const float* b1       = (const float*)d_in[4];
    const float* g1       = (const float*)d_in[5];
    const float* beta1    = (const float*)d_in[6];
    const float* m1       = (const float*)d_in[7];
    const float* v1       = (const float*)d_in[8];
    const float* w2       = (const float*)d_in[9];
    const float* b2       = (const float*)d_in[10];
    const float* g2       = (const float*)d_in[11];
    const float* beta2    = (const float*)d_in[12];
    const float* m2       = (const float*)d_in[13];
    const float* v2       = (const float*)d_in[14];
    const float* fc1_w    = (const float*)d_in[15];
    const float* fc1_b    = (const float*)d_in[16];
    const float* fc2_w    = (const float*)d_in[17];
    const float* fc2_b    = (const float*)d_in[18];
    float*       out      = (float*)d_out;

    char* ws = (char*)d_ws;
    _Float16* w1e  = (_Float16*)(ws + WS_W1E);
    _Float16* w2e  = (_Float16*)(ws + WS_W2E);
    _Float16* fc1e = (_Float16*)(ws + WS_FC1E);
    float*    bnp  = (float*)(ws + WS_BNP);

    // Kernel 0: weight repack (cheap one-time per launch)
    {
        int total  = 64 * 160 + 64 * 256 + 64 * 5184 + 256;
        int blocks = (total + 255) / 256;
        jung_repack<<<blocks, 256, 0, stream>>>(w1, w2, fc1_w,
                                                b1, g1, beta1, m1, v1,
                                                b2, g2, beta2, m2, v2,
                                                w1e, w2e, fc1e, bnp);
    }

    // Kernel 1: fused decoder, 8 samples per workgroup
    {
        const int B = 32768;
        jung_fused<<<B / SB, NTHREADS, SMEM_BYTES, stream>>>(
            syndrome, Hz, Hx, w1e, w2e, fc1e, bnp, fc1_b, fc2_w, fc2_b, out);
    }
}
